// TransformerLayer_21973052686445
// MI455X (gfx1250) — compile-verified, matrix operand order checked by reading, not run on hardware
//
#include <hip/hip_runtime.h>

typedef __bf16 bf16_t;
typedef __attribute__((ext_vector_type(8)))  __bf16 v8bf;
typedef __attribute__((ext_vector_type(16))) __bf16 v16bf;
typedef __attribute__((ext_vector_type(8)))  float  v8f;
typedef __attribute__((address_space(3))) char lds_char;

#define BATCH 4
#define SEQL  1024
#define HID   2048
#define NHEAD 16
#define DHEAD 128
#define NTOK  (BATCH * SEQL)          // 4096 tokens

// LDS row padding (bank-conflict avoidance: 64 banks x 4B = 256B stripe)
#define AST 136                        // staged A row stride (elems): 272B = 17*16B
#define SST 1032                       // score/prob row stride (elems): != 0 mod 256B

// unit-scaling constants
#define S_FANH      0.022097086912079608f   // 2048^-0.5
#define S_FAN4H     0.011048543456039804f   // 8192^-0.5
#define S_SCORE     0.08838834764831845f    // 128^-0.5
#define S_PROB      19.408981098382773f     // sqrt(1024/e): softmax*(s/sqrt(e)) * s^-0.5
#define INV_GELURMS 1.5335306f              // 1/sqrt(1/3 + 1/(2*pi*sqrt(3)))
#define RA_MHSA     0.1f                    // sqrt(0.01)
#define RB_MHSA     0.99498743710662f       // sqrt(0.99)
#define RA_MLP      0.7071067811865476f     // sqrt(0.5)

enum { EPI_QKV = 0, EPI_OPROJ = 1, EPI_MLP1 = 2, EPI_MLP2 = 3 };

__device__ __forceinline__ v8f wmma_bf16f32(v16bf a, v16bf b, v8f c) {
  // (neg_a, A, neg_b, B, c_mod, C, reuse_a, reuse_b)
  return __builtin_amdgcn_wmma_f32_16x16x32_bf16(false, a, false, b, (short)0, c, false, false);
}

// Load a 16x32 bf16 WMMA operand fragment (A or B) for this lane.
// row_ptr points at the start of this lane's row. ISA 7.12.2 layout:
// element e holds K = (e>=8?16:0) + (lane>=16?8:0) + (e&7)  ->  two 16B chunks.
__device__ __forceinline__ v16bf load_frag(const bf16_t* row_ptr, int k0, int lane) {
  const int hi8 = (lane >> 4) << 3;                       // 0 or 8
  v8bf c0 = *(const v8bf*)(row_ptr + k0 + hi8);
  v8bf c1 = *(const v8bf*)(row_ptr + k0 + 16 + hi8);
  return __builtin_shufflevector(c0, c1, 0,1,2,3,4,5,6,7,8,9,10,11,12,13,14,15);
}

// async global -> LDS, 16 bytes per lane; tracked by ASYNCcnt
__device__ __forceinline__ void async_load_b128(const bf16_t* gaddr, bf16_t* lds_generic) {
  unsigned lofs = (unsigned)(unsigned long)(lds_char*)lds_generic;  // LDS byte offset
  asm volatile("global_load_async_to_lds_b128 %0, %1, off"
               :: "v"(lofs), "v"(gaddr) : "memory");
}
__device__ __forceinline__ void wait_asynccnt0() {
  asm volatile("s_wait_asynccnt 0x0" ::: "memory");
}

// ---------------------------------------------------------------- fp32 -> bf16
__global__ void f2bf_kernel(const float* __restrict__ in, bf16_t* __restrict__ out, int n) {
  int i = blockIdx.x * blockDim.x + threadIdx.x;
  int stride = gridDim.x * blockDim.x;
  for (; i < n; i += stride) out[i] = (bf16_t)in[i];
}

// ---------------------------------------------------------------- LayerNorm -> bf16
__global__ void __launch_bounds__(256)
ln_bf16_kernel(const float* __restrict__ x, const float* __restrict__ w,
               const float* __restrict__ b, bf16_t* __restrict__ y) {
  const int row = blockIdx.x;
  const int tid = threadIdx.x;
  const float* xr = x + (size_t)row * HID;
  float s = 0.f, s2 = 0.f;
  for (int i = tid; i < HID; i += 256) { float v = xr[i]; s += v; s2 += v * v; }
  __shared__ float rs[256], rs2[256];
  rs[tid] = s; rs2[tid] = s2; __syncthreads();
  for (int off = 128; off > 0; off >>= 1) {
    if (tid < off) { rs[tid] += rs[tid + off]; rs2[tid] += rs2[tid + off]; }
    __syncthreads();
  }
  const float mean = rs[0] * (1.0f / HID);
  const float var  = rs2[0] * (1.0f / HID) - mean * mean;
  const float inv  = rsqrtf(var + 1e-5f);
  for (int i = tid; i < HID; i += 256) {
    float v = (xr[i] - mean) * inv * w[i] + b[i];
    y[(size_t)row * HID + i] = (bf16_t)v;
  }
}

// ---------------------------------------------------------------- WMMA GEMM
// C[M,N] = A[M,K](bf16, row-major) * W[N,K](bf16, row-major)^T, fp32 accum.
// Block: 256 threads = 8 waves arranged 2(M) x 4(N); block tile 64(M) x 256(N).
// Wave tile: 32(M) x 64(N) = 8 accumulators.
// A block tile is staged K=128 deep (64x128 bf16 = 16KB, padded rows) and
// double-buffered via global_load_async_to_lds_b128 (ASYNCcnt): one
// s_wait_asynccnt + barrier per 32 WMMAs, copy fully overlapped with compute.
template <int EPI>
__global__ void __launch_bounds__(256)
gemm_bf16_kernel(const bf16_t* __restrict__ A, const bf16_t* __restrict__ W,
                 int N, int K, float scale,
                 const float* __restrict__ bias,
                 const float* __restrict__ resid, float ra, float rb,
                 float* __restrict__ outF, bf16_t* __restrict__ outH,
                 bf16_t* __restrict__ qOut, bf16_t* __restrict__ kOut,
                 bf16_t* __restrict__ vTOut) {
  __shared__ bf16_t sA[2][64 * AST];   // two padded 64x128 A-tile buffers

  const int tid  = threadIdx.x;
  const int wave = tid >> 5;
  const int wm   = wave >> 2;        // 0..1
  const int wn   = wave & 3;         // 0..3
  const int lane = tid & 31;
  const int lrow = lane & 15;
  const int hi   = lane >> 4;
  const int m_blk = blockIdx.y * 64;
  const int m0 = m_blk + wm * 32;
  const int n0 = blockIdx.x * 256 + wn * 64;

  const bf16_t* wrow0 = W + (size_t)(n0 +  0 + lrow) * K;
  const bf16_t* wrow1 = W + (size_t)(n0 + 16 + lrow) * K;
  const bf16_t* wrow2 = W + (size_t)(n0 + 32 + lrow) * K;
  const bf16_t* wrow3 = W + (size_t)(n0 + 48 + lrow) * K;

  // cooperative staging of one 64x128 tile: 1024 16B chunks / 256 threads = 4 each
  auto stage = [&](int buf, int k0) {
#pragma unroll
    for (int i = 0; i < 4; i++) {
      const int c   = tid + 256 * i;
      const int row = c >> 4;              // 16 chunks per 128-elem row
      const int col = (c & 15) * 8;
      async_load_b128(A + (size_t)(m_blk + row) * K + k0 + col,
                      &sA[buf][row * AST + col]);
    }
  };

  v8f acc[2][4];
#pragma unroll
  for (int mi = 0; mi < 2; mi++)
#pragma unroll
    for (int j = 0; j < 4; j++)
#pragma unroll
      for (int r = 0; r < 8; r++) acc[mi][j][r] = 0.f;

  // prime buffer 0
  stage(0, 0);
  wait_asynccnt0();
  __syncthreads();

  for (int k0 = 0; k0 < K; k0 += 128) {
    const int  cur  = (k0 >> 7) & 1;
    const bool more = (k0 + 128) < K;
    if (more) stage(cur ^ 1, k0 + 128);      // overlaps the 32 WMMAs below

    if (more) {  // prefetch next stage's weight rows (global_prefetch_b8)
      __builtin_prefetch(wrow0 + k0 + 128, 0, 1);
      __builtin_prefetch(wrow1 + k0 + 128, 0, 1);
      __builtin_prefetch(wrow2 + k0 + 128, 0, 1);
      __builtin_prefetch(wrow3 + k0 + 128, 0, 1);
    }

    const bf16_t* aR0 = sA[cur] + (wm * 32 + lrow) * AST;
    const bf16_t* aR1 = sA[cur] + (wm * 32 + 16 + lrow) * AST;
#pragma unroll
    for (int ks = 0; ks < 128; ks += 32) {
      v16bf a0 = load_frag(aR0, ks, lane);   // ds_load_b128, bank-spread rows
      v16bf a1 = load_frag(aR1, ks, lane);
      v16bf b0 = load_frag(wrow0, k0 + ks, lane);
      v16bf b1 = load_frag(wrow1, k0 + ks, lane);
      v16bf b2 = load_frag(wrow2, k0 + ks, lane);
      v16bf b3 = load_frag(wrow3, k0 + ks, lane);
      acc[0][0] = wmma_bf16f32(a0, b0, acc[0][0]);
      acc[1][0] = wmma_bf16f32(a1, b0, acc[1][0]);
      acc[0][1] = wmma_bf16f32(a0, b1, acc[0][1]);
      acc[1][1] = wmma_bf16f32(a1, b1, acc[1][1]);
      acc[0][2] = wmma_bf16f32(a0, b2, acc[0][2]);
      acc[1][2] = wmma_bf16f32(a1, b2, acc[1][2]);
      acc[0][3] = wmma_bf16f32(a0, b3, acc[0][3]);
      acc[1][3] = wmma_bf16f32(a1, b3, acc[1][3]);
    }

    if (more) wait_asynccnt0();   // my wave's async copy landed
    __syncthreads();              // block-wide: next buffer valid, cur buffer free
  }

#pragma unroll
  for (int mi = 0; mi < 2; mi++) {
#pragma unroll
    for (int j = 0; j < 4; j++) {
      const int ncol = n0 + j * 16 + lrow;
#pragma unroll
      for (int r = 0; r < 8; r++) {
        const int m = m0 + mi * 16 + r + 8 * hi;  // D layout: col=lane&15, row=r+8*(lane>>4)
        float v = acc[mi][j][r] * scale;
        if (EPI == EPI_QKV) {
          const int z = ncol >> 11, rem = ncol & 2047;
          const int head = rem >> 7, dd = rem & 127;
          const size_t bh = (size_t)(m >> 10) * NHEAD + head;
          const int sr = m & (SEQL - 1);
          if (z == 0)      qOut [(bh * SEQL + sr) * DHEAD + dd] = (bf16_t)v;
          else if (z == 1) kOut [(bh * SEQL + sr) * DHEAD + dd] = (bf16_t)v;
          else             vTOut[(bh * DHEAD + dd) * SEQL + sr] = (bf16_t)v;  // V transposed
        } else if (EPI == EPI_OPROJ) {
          const size_t idx = (size_t)m * N + ncol;
          outF[idx] = v * ra + resid[idx] * rb;
        } else if (EPI == EPI_MLP1) {
          const float u = v + bias[ncol];
          const float g = u * (0.5f * (1.0f + erff(u * 0.70710678f))) * INV_GELURMS;
          outH[(size_t)m * N + ncol] = (bf16_t)g;
        } else {  // EPI_MLP2
          const float u = v + bias[ncol];
          const size_t idx = (size_t)m * N + ncol;
          outF[idx] = u * ra + resid[idx] * rb;
        }
      }
    }
  }
}

// ---------------------------------------------------------------- Attention
// One block (128 thr = 4 waves) per (b, h, 16-row tile). Scores into LDS (fp32),
// two-pass softmax, bf16 probs into LDS, then probs @ V^T via WMMA.
__global__ void __launch_bounds__(128)
attn_kernel(const bf16_t* __restrict__ q, const bf16_t* __restrict__ k,
            const bf16_t* __restrict__ vT, bf16_t* __restrict__ ctx) {
  extern __shared__ char smem[];
  float*  sS   = (float*)smem;                               // 16 x SST fp32 (padded)
  bf16_t* sP   = (bf16_t*)(smem + 16 * SST * 4);             // 16 x SST bf16 (padded)
  float*  sRed = (float*)(smem + 16 * SST * 4 + 16 * SST * 2);  // 16 x 8

  const int bh = blockIdx.y;
  const int b  = bh >> 4;
  const int h  = bh & 15;
  const int m0 = blockIdx.x * 16;
  const int tid  = threadIdx.x;
  const int wave = tid >> 5;
  const int lane = tid & 31;
  const int lrow = lane & 15;
  const int hi   = lane >> 4;

  const bf16_t* qbase = q  + (size_t)bh * SEQL * DHEAD;
  const bf16_t* kbase = k  + (size_t)bh * SEQL * DHEAD;
  const bf16_t* vbase = vT + (size_t)bh * DHEAD * SEQL;

  // Q fragments for this row tile (shared by all t-tiles this wave computes)
  v16bf qa[4];
#pragma unroll
  for (int j = 0; j < 4; j++)
    qa[j] = load_frag(qbase + (size_t)(m0 + lrow) * DHEAD, j * 32, lane);

  // scores = (q . k) * d^-0.5 ; wave-strided over 64 t-tiles
  for (int ti = wave; ti < SEQL / 16; ti += 4) {
    const int t0 = ti * 16;
    v8f acc;
#pragma unroll
    for (int r = 0; r < 8; r++) acc[r] = 0.f;
#pragma unroll
    for (int j = 0; j < 4; j++) {
      v16bf kb = load_frag(kbase + (size_t)(t0 + lrow) * DHEAD, j * 32, lane);
      acc = wmma_bf16f32(qa[j], kb, acc);
    }
#pragma unroll
    for (int r = 0; r < 8; r++)
      sS[(r + 8 * hi) * SST + t0 + lrow] = acc[r] * S_SCORE;
  }
  __syncthreads();

  // softmax over each of 16 rows; 8 threads per row, interleaved columns
  // (c = sub + 8*i -> consecutive banks across the 8 sub-threads of a row)
  {
    const int row = tid >> 3, sub = tid & 7;
    float* srow = sS + row * SST;
    float mx = -1e30f;
    for (int i = 0; i < 128; i++) mx = fmaxf(mx, srow[sub + 8 * i]);
    sRed[row * 8 + sub] = mx;
    __syncthreads();
    mx = -1e30f;
#pragma unroll
    for (int i = 0; i < 8; i++) mx = fmaxf(mx, sRed[row * 8 + i]);
    __syncthreads();
    float sm = 0.f;
    for (int i = 0; i < 128; i++) sm += __expf(srow[sub + 8 * i] - mx);
    sRed[row * 8 + sub] = sm;
    __syncthreads();
    sm = 0.f;
#pragma unroll
    for (int i = 0; i < 8; i++) sm += sRed[row * 8 + i];
    const float sc = S_PROB / sm;   // folds softmax norm, s/sqrt(e), and s^-0.5
    for (int i = 0; i < 128; i++) {
      const int c = sub + 8 * i;
      sP[row * SST + c] = (bf16_t)(__expf(srow[c] - mx) * sc);
    }
  }
  __syncthreads();

  // ctx = probs @ V : contraction over t (K=1024), wave owns d-tiles wave*32 + {0,16}
  v8f acc0, acc1;
#pragma unroll
  for (int r = 0; r < 8; r++) { acc0[r] = 0.f; acc1[r] = 0.f; }
  const bf16_t* prow  = sP + (size_t)lrow * SST;
  const bf16_t* v0row = vbase + (size_t)(wave * 32 + lrow) * SEQL;
  const bf16_t* v1row = vbase + (size_t)(wave * 32 + 16 + lrow) * SEQL;
  for (int kk = 0; kk < SEQL; kk += 32) {
    v16bf pa  = load_frag(prow,  kk, lane);      // LDS -> ds_load_b128, bank-spread
    v16bf bv0 = load_frag(v0row, kk, lane);
    v16bf bv1 = load_frag(v1row, kk, lane);
    acc0 = wmma_bf16f32(pa, bv0, acc0);
    acc1 = wmma_bf16f32(pa, bv1, acc1);
  }
#pragma unroll
  for (int r = 0; r < 8; r++) {
    const int m = m0 + r + 8 * hi;
    const size_t base = ((size_t)(b * SEQL + m)) * HID + h * DHEAD;
    ctx[base + wave * 32 + lrow]      = (bf16_t)acc0[r];
    ctx[base + wave * 32 + 16 + lrow] = (bf16_t)acc1[r];
  }
}

// ---------------------------------------------------------------- launch
extern "C" void kernel_launch(void* const* d_in, const int* in_sizes, int n_in,
                              void* d_out, int out_size, void* d_ws, size_t ws_size,
                              hipStream_t stream) {
  const float* x     = (const float*)d_in[0];
  const float* ln1_w = (const float*)d_in[1];
  const float* ln1_b = (const float*)d_in[2];
  const float* w_qkv = (const float*)d_in[3];
  const float* w_o   = (const float*)d_in[4];
  const float* ln2_w = (const float*)d_in[5];
  const float* ln2_b = (const float*)d_in[6];
  const float* w1    = (const float*)d_in[7];
  const float* b1    = (const float*)d_in[8];
  const float* w2    = (const float*)d_in[9];
  const float* b2    = (const float*)d_in[10];

  char* p = (char*)d_ws;
  auto take = [&](size_t bytes) -> char* {
    char* r = p;
    p += (bytes + 255) & ~(size_t)255;
    return r;
  };
  bf16_t* yb    = (bf16_t*)take((size_t)NTOK * HID * 2);
  bf16_t* wqkvb = (bf16_t*)take((size_t)3 * HID * HID * 2);
  bf16_t* wob   = (bf16_t*)take((size_t)HID * HID * 2);
  bf16_t* w1b   = (bf16_t*)take((size_t)4 * HID * HID * 2);
  bf16_t* w2b   = (bf16_t*)take((size_t)4 * HID * HID * 2);
  bf16_t* qb    = (bf16_t*)take((size_t)BATCH * NHEAD * SEQL * DHEAD * 2);
  bf16_t* kb    = (bf16_t*)take((size_t)BATCH * NHEAD * SEQL * DHEAD * 2);
  bf16_t* vtb   = (bf16_t*)take((size_t)BATCH * NHEAD * SEQL * DHEAD * 2);
  bf16_t* ctxb  = (bf16_t*)take((size_t)NTOK * HID * 2);
  float*  x2    = (float*) take((size_t)NTOK * HID * 4);
  bf16_t* y2b   = (bf16_t*)take((size_t)NTOK * HID * 2);
  bf16_t* h1b   = (bf16_t*)take((size_t)NTOK * 4 * HID * 2);

  // 1) weights fp32 -> bf16 (one-time per call; ~300 MB @ 23.3 TB/s)
  f2bf_kernel<<<2048, 256, 0, stream>>>(w_qkv, wqkvb, 3 * HID * HID);
  f2bf_kernel<<<2048, 256, 0, stream>>>(w_o,   wob,   HID * HID);
  f2bf_kernel<<<2048, 256, 0, stream>>>(w1,    w1b,   4 * HID * HID);
  f2bf_kernel<<<2048, 256, 0, stream>>>(w2,    w2b,   4 * HID * HID);

  // 2) LN1
  ln_bf16_kernel<<<NTOK, 256, 0, stream>>>(x, ln1_w, ln1_b, yb);

  // 3) QKV GEMM (M=4096, N=6144, K=2048), scatter q/k/[v^T]
  gemm_bf16_kernel<EPI_QKV><<<dim3(3 * HID / 256, NTOK / 64), 256, 0, stream>>>(
      yb, wqkvb, 3 * HID, HID, S_FANH, nullptr, nullptr, 0.f, 0.f,
      nullptr, nullptr, qb, kb, vtb);

  // 4) attention (grid: 64 row tiles x 64 batch-heads), ~97 KB dynamic LDS
  attn_kernel<<<dim3(SEQL / 16, BATCH * NHEAD), 128,
                16 * SST * 4 + 16 * SST * 2 + 16 * 8 * 4, stream>>>(qb, kb, vtb, ctxb);

  // 5) O-proj + MHSA residual -> x2 (fp32)
  gemm_bf16_kernel<EPI_OPROJ><<<dim3(HID / 256, NTOK / 64), 256, 0, stream>>>(
      ctxb, wob, HID, HID, S_FANH, nullptr, x, RA_MHSA, RB_MHSA,
      x2, nullptr, nullptr, nullptr, nullptr);

  // 6) LN2 + MLP up-proj + unit-scaled GELU -> bf16
  ln_bf16_kernel<<<NTOK, 256, 0, stream>>>(x2, ln2_w, ln2_b, y2b);
  gemm_bf16_kernel<EPI_MLP1><<<dim3(4 * HID / 256, NTOK / 64), 256, 0, stream>>>(
      y2b, w1b, 4 * HID, HID, S_FANH, b1, nullptr, 0.f, 0.f,
      nullptr, h1b, nullptr, nullptr, nullptr);

  // 7) MLP down-proj + bias + MLP residual -> d_out (fp32)
  gemm_bf16_kernel<EPI_MLP2><<<dim3(HID / 256, NTOK / 64), 256, 0, stream>>>(
      h1b, w2b, HID, 4 * HID, S_FAN4H, b2, x2, RA_MLP, RA_MLP,
      (float*)d_out, nullptr, nullptr, nullptr, nullptr);
}
